// DynamicGraphCNN_87050397155552
// MI455X (gfx1250) — matrix operand order repaired and checked by reading, not compiled, not run on hardware
//
#include <hip/hip_runtime.h>

typedef _Float16 half16 __attribute__((ext_vector_type(16)));
typedef float    float8 __attribute__((ext_vector_type(8)));

#define B_     8
#define N_     2048
#define KNN_   20
#define SLOPE_ 0.2f

// ---------------- helpers ----------------

// monotone float <-> signed-int key (for atomicMax on floats of any sign)
__device__ __forceinline__ int fkey(float f){
  int u = __float_as_int(f);
  return (u >= 0) ? u : (u ^ 0x7FFFFFFF);
}
__device__ __forceinline__ float finv(int k){
  return __int_as_float((k >= 0) ? k : (k ^ 0x7FFFFFFF));
}
// monotone float -> unsigned key (for 64-bit packed min-reduction)
__device__ __forceinline__ unsigned fkey_u(float f){
  unsigned u = __float_as_uint(f);
  return (u & 0x80000000u) ? ~u : (u | 0x80000000u);
}
__device__ __forceinline__ float lrelu(float x){ return x > 0.f ? x : SLOPE_ * x; }

// A fragment from pre-swizzled f16 weights: tile tIdx, lane's 16 halves are 32
// contiguous bytes -> two global_load_b128, no conversions in the hot loop.
__device__ __forceinline__ half16 load_a_packed(const _Float16* __restrict__ P, int tIdx, int lane){
  const int4* q = (const int4*)(P + ((size_t)tIdx * 32 + lane) * 16);
  half16 a;
  ((int4*)&a)[0] = q[0];
  ((int4*)&a)[1] = q[1];
  return a;
}

// B fragment: 32x16 f16 tile from LDS, column-major E[col][ldk].
// lanes 0-15: col=lane, K = kb+0..15 ; lanes 16-31: col=lane-16, K = kb+16..31
// 16 consecutive halves, 16B aligned -> ds_load_b128 x2.
__device__ __forceinline__ half16 load_b_lds(const _Float16* E, int ldk, int col, int kb, int lane){
  const int4* q = (const int4*)(E + (size_t)col * ldk + kb + ((lane >> 4) << 4));
  half16 bf;
  ((int4*)&bf)[0] = q[0];
  ((int4*)&bf)[1] = q[1];
  return bf;
}

// ---------------- kernels ----------------

// Pack fp32 weights W[M][Kdim] into fragment-major f16:
// out[((mt*(KP/32)+kt)*32 + lane)*16 + e] = W[mt*16 + (lane&15)][kt*32 + kpos(lane,e)]
// kpos = (lane>>4)*16 + e; zero-padded past Kdim.
__global__ __launch_bounds__(256) void pack_a_kernel(const float* __restrict__ W,
                                                     int M, int Kdim, int KP,
                                                     _Float16* __restrict__ out){
  int t = blockIdx.x * blockDim.x + threadIdx.x;
  if (t >= M * KP) return;
  int e    = t & 15;
  int lane = (t >> 4) & 31;
  int tIdx = t >> 9;
  int KT   = KP >> 5;
  int mt   = tIdx / KT, kt = tIdx % KT;
  int row  = mt * 16 + (lane & 15);
  int k    = kt * 32 + ((lane >> 4) << 4) + e;
  float v  = (k < Kdim) ? W[(size_t)row * Kdim + k] : 0.f;
  out[t] = (_Float16)v;
}

// (B,3,N) -> (B,N,3)
__global__ __launch_bounds__(256) void transpose_points_kernel(const float* __restrict__ pts,
                                                               float* __restrict__ out){
  int t = blockIdx.x * blockDim.x + threadIdx.x;
  if (t >= B_ * N_) return;
  int b = t / N_, n = t % N_;
  #pragma unroll
  for (int c = 0; c < 3; ++c)
    out[(size_t)t * 3 + c] = pts[((size_t)b * 3 + c) * N_ + n];
}

// kNN: one wave per point. Per-lane sorted top-20 kept in registers (static
// indices only -> no scratch), then 20 rounds of wave-wide packed-u64 min merge.
__global__ __launch_bounds__(256) void knn_kernel(const float* __restrict__ xt, int ld, int C,
                                                  int* __restrict__ idx_out){
  __shared__ float cen[8][128];
  const int tid = threadIdx.x, wave = tid >> 5, lane = tid & 31;
  const int b = blockIdx.y;
  const int n = blockIdx.x * 8 + wave;
  const float* crow = xt + ((size_t)b * N_ + n) * ld;
  for (int k = lane; k < C; k += 32) cen[wave][k] = crow[k];
  __syncthreads();

  float bd[KNN_]; int bi[KNN_];
  #pragma unroll
  for (int i = 0; i < KNN_; ++i){ bd[i] = 3.0e38f; bi[i] = 0x7FFFFFFF; }

  const bool vec4 = ((C & 3) == 0) && ((ld & 3) == 0);
  for (int m = lane; m < N_; m += 32){
    const float* row = xt + ((size_t)b * N_ + m) * ld;
    float d = 0.f;
    if (vec4){
      const float4* r4 = (const float4*)row;
      const float4* c4 = (const float4*)&cen[wave][0];
      for (int k = 0; k < (C >> 2); ++k){
        float4 rv = r4[k], cv = c4[k];
        float t0 = cv.x - rv.x, t1 = cv.y - rv.y, t2 = cv.z - rv.z, t3 = cv.w - rv.w;
        d = fmaf(t0, t0, d); d = fmaf(t1, t1, d);
        d = fmaf(t2, t2, d); d = fmaf(t3, t3, d);
      }
    } else {
      for (int k = 0; k < C; ++k){ float t = cen[wave][k] - row[k]; d = fmaf(t, t, d); }
    }
    if (d < bd[KNN_ - 1]){
      bd[KNN_ - 1] = d; bi[KNN_ - 1] = m;
      #pragma unroll
      for (int i = KNN_ - 1; i > 0; --i){
        if (bd[i] < bd[i - 1]){
          float td = bd[i]; bd[i] = bd[i - 1]; bd[i - 1] = td;
          int   ti = bi[i]; bi[i] = bi[i - 1]; bi[i - 1] = ti;
        }
      }
    }
  }

  int* orow = idx_out + ((size_t)b * N_ + n) * KNN_;
  for (int r = 0; r < KNN_; ++r){
    unsigned long long key = ((unsigned long long)fkey_u(bd[0]) << 32) | (unsigned)bi[0];
    unsigned long long w = key;
    #pragma unroll
    for (int mm = 1; mm < 32; mm <<= 1){
      unsigned lo = (unsigned)__shfl_xor((int)(unsigned)w, mm, 32);
      unsigned hi = (unsigned)__shfl_xor((int)(unsigned)(w >> 32), mm, 32);
      unsigned long long o = ((unsigned long long)hi << 32) | lo;
      if (o < w) w = o;
    }
    if (lane == 0) orow[r] = (int)(w & 0xFFFFFFFFull);
    if (key == w){                       // winning lane pops its head (static shift)
      #pragma unroll
      for (int i = 0; i < KNN_ - 1; ++i){ bd[i] = bd[i + 1]; bi[i] = bi[i + 1]; }
      bd[KNN_ - 1] = 3.0e38f; bi[KNN_ - 1] = 0x7FFFFFFF;
    }
  }
}

// EdgeConv: y = max_k lrelu(bn(W @ [nb-c; c])). One WG = (batch, 16-point tile).
// Edge matrix (KP x 320) staged f16 column-major in dynamic LDS (up to 160KB on CDNA5).
// A fragments (pre-swizzled f16) hoisted per M-tile and reused across 10-20 column tiles.
template<int CIN, int COUT, int KP>
__global__ __launch_bounds__(256) void edgeconv_kernel(
    const float* __restrict__ xin, int ldi,
    const int*   __restrict__ idx,
    const _Float16* __restrict__ wp,       // packed weights, (COUT/16)*(KP/32) tiles
    const float* __restrict__ gg, const float* __restrict__ bb,
    float* __restrict__ out, int ldo){
  extern __shared__ char smem[];
  _Float16* E  = (_Float16*)smem;                         // [320][KP] col-major
  int* omax    = (int*)(smem + (size_t)320 * KP * 2);     // [16][COUT] monotone keys
  const int tid = threadIdx.x, wave = tid >> 5, lane = tid & 31;
  const int b = blockIdx.y, n0 = blockIdx.x * 16;

  for (int col = tid; col < 16 * KNN_; col += 256){
    int p = col / KNN_, j = col % KNN_;
    int m = idx[((size_t)b * N_ + n0 + p) * KNN_ + j];
    const float* ce = xin + ((size_t)b * N_ + n0 + p) * ldi;
    const float* nb = xin + ((size_t)b * N_ + m) * ldi;
    _Float16* ec = E + (size_t)col * KP;
    for (int k = 0; k < CIN; ++k)       ec[k]       = (_Float16)(nb[k] - ce[k]);
    for (int k = 0; k < CIN; ++k)       ec[CIN + k] = (_Float16)ce[k];
    for (int k = 2 * CIN; k < KP; ++k)  ec[k]       = (_Float16)0.f;
  }
  for (int i = tid; i < 16 * COUT; i += 256) omax[i] = 0x80000000;
  __syncthreads();

  const float bnmul = rsqrtf(1.0f + 1e-5f);
  constexpr int MT  = COUT / 16;
  constexpr int KT  = KP / 32;
  constexpr int WPM = (MT >= 8) ? 1 : (8 / MT);   // waves cooperating on one M-tile
  constexpr int CTN = KNN_ / WPM;                 // column tiles per wave (20 or 10)
  const int ctBase  = (wave % WPM) * CTN;

  for (int mt = wave / WPM; mt < MT; mt += 8 / WPM){
    half16 afr[KT];
    #pragma unroll
    for (int kt = 0; kt < KT; ++kt) afr[kt] = load_a_packed(wp, mt * KT + kt, lane);
    for (int ci = 0; ci < CTN; ++ci){
      int ct = ctBase + ci;
      float8 acc = {};
      #pragma unroll
      for (int kt = 0; kt < KT; ++kt){
        half16 bf = load_b_lds(E, KP, ct * 16 + (lane & 15), kt * 32, lane);
        acc = __builtin_amdgcn_wmma_f32_16x16x32_f16(false, afr[kt], false, bf,
                                                     (short)0, acc, false, false);
      }
      #pragma unroll
      for (int i = 0; i < 8; ++i){
        int M   = mt * 16 + i + ((lane >> 4) << 3);
        int col = ct * 16 + (lane & 15);
        int p   = col / KNN_;
        float v = lrelu(acc[i] * (gg[M] * bnmul) + bb[M]);
        atomicMax(&omax[p * COUT + M], fkey(v));   // ds_max_i32, fused max-over-k
      }
    }
  }
  __syncthreads();
  for (int i = tid; i < 16 * COUT; i += 256){
    int p = i / COUT, ch = i % COUT;
    out[((size_t)b * N_ + n0 + p) * ldo + ch] = finv(omax[i]);
  }
}

__global__ __launch_bounds__(256) void init_pool_kernel(int* maxbuf, float* sumbuf){
  int t = blockIdx.x * blockDim.x + threadIdx.x;
  if (t < B_ * 1024){ maxbuf[t] = 0x80000000; sumbuf[t] = 0.f; }
}

// 1024x512 pointwise MLP fused with global max/mean pool (activation never hits HBM).
__global__ __launch_bounds__(256) void mlp_pool_kernel(
    const float* __restrict__ feat,
    const _Float16* __restrict__ wp,               // packed 1024x512 weights
    const float* __restrict__ gg, const float* __restrict__ bb,
    int* __restrict__ maxbuf, float* __restrict__ sumbuf){
  __shared__ _Float16 Bs[16 * 512];
  const int tid = threadIdx.x, wave = tid >> 5, lane = tid & 31;
  const int b = blockIdx.y, n0 = blockIdx.x * 16;
  for (int i = tid; i < 16 * 512; i += 256){
    int p = i >> 9, k = i & 511;
    Bs[i] = (_Float16)feat[((size_t)b * N_ + n0 + p) * 512 + k];
  }
  __syncthreads();
  const float bnmul = rsqrtf(1.0f + 1e-5f);
  for (int mt = wave; mt < 64; mt += 8){
    float8 acc = {};
    #pragma unroll
    for (int kt = 0; kt < 16; ++kt){
      half16 a  = load_a_packed(wp, mt * 16 + kt, lane);
      half16 bf = load_b_lds(Bs, 512, lane & 15, kt * 32, lane);
      acc = __builtin_amdgcn_wmma_f32_16x16x32_f16(false, a, false, bf,
                                                   (short)0, acc, false, false);
    }
    #pragma unroll
    for (int i = 0; i < 8; ++i){
      int ch  = mt * 16 + i + ((lane >> 4) << 3);
      float v = lrelu(acc[i] * (gg[ch] * bnmul) + bb[ch]);
      float mx = v, sm = v;
      #pragma unroll
      for (int mm = 1; mm < 16; mm <<= 1){           // reduce across 16 points in-register
        mx = fmaxf(mx, __shfl_xor(mx, mm, 32));
        sm += __shfl_xor(sm, mm, 32);
      }
      if ((lane & 15) == 0){
        atomicMax(&maxbuf[b * 1024 + ch], fkey(mx));
        atomicAdd(&sumbuf[b * 1024 + ch], sm);
      }
    }
  }
}

// FC head: tiny (8x2048 @ 2048x512 ...), single WG.
__global__ __launch_bounds__(512) void head_kernel(
    const int* __restrict__ maxbuf, const float* __restrict__ sumbuf,
    const float* __restrict__ fc1w, const float* __restrict__ fc1bias,
    const float* __restrict__ fc1g, const float* __restrict__ fc1b,
    const float* __restrict__ fc2w, const float* __restrict__ fc2bias,
    const float* __restrict__ fc2g, const float* __restrict__ fc2b,
    const float* __restrict__ fc3w, const float* __restrict__ fc3bias,
    float* __restrict__ outp){
  __shared__ float f2[2048], h1[512], h2[256];
  const int tid = threadIdx.x;
  const float bnmul = rsqrtf(1.0f + 1e-5f);
  for (int b = 0; b < B_; ++b){
    for (int i = tid; i < 1024; i += 512){
      f2[i]        = finv(maxbuf[b * 1024 + i]);
      f2[1024 + i] = sumbuf[b * 1024 + i] * (1.0f / N_);
    }
    __syncthreads();
    { int o = tid;
      float s = fc1bias[o];
      const float* wr = fc1w + (size_t)o * 2048;
      for (int k = 0; k < 2048; ++k) s = fmaf(f2[k], wr[k], s);
      h1[o] = lrelu(s * (fc1g[o] * bnmul) + fc1b[o]);
    }
    __syncthreads();
    if (tid < 256){ int o = tid;
      float s = fc2bias[o];
      const float* wr = fc2w + (size_t)o * 512;
      for (int k = 0; k < 512; ++k) s = fmaf(h1[k], wr[k], s);
      h2[o] = lrelu(s * (fc2g[o] * bnmul) + fc2b[o]);
    }
    __syncthreads();
    if (tid < 40){ int o = tid;
      float s = fc3bias[o];
      const float* wr = fc3w + (size_t)o * 256;
      for (int k = 0; k < 256; ++k) s = fmaf(h2[k], wr[k], s);
      outp[b * 40 + o] = s;
    }
    __syncthreads();
  }
}

// ---------------- launch ----------------

extern "C" void kernel_launch(void* const* d_in, const int* in_sizes, int n_in,
                              void* d_out, int out_size, void* d_ws, size_t ws_size,
                              hipStream_t stream){
  const float* points   = (const float*)d_in[0];
  const float* ec1_w    = (const float*)d_in[1];
  const float* ec1_g    = (const float*)d_in[2];
  const float* ec1_b    = (const float*)d_in[3];
  const float* ec2_w    = (const float*)d_in[4];
  const float* ec2_g    = (const float*)d_in[5];
  const float* ec2_b    = (const float*)d_in[6];
  const float* ec3_w    = (const float*)d_in[7];
  const float* ec3_g    = (const float*)d_in[8];
  const float* ec3_b    = (const float*)d_in[9];
  const float* ec4_w    = (const float*)d_in[10];
  const float* ec4_g    = (const float*)d_in[11];
  const float* ec4_b    = (const float*)d_in[12];
  const float* mlp_w    = (const float*)d_in[13];
  const float* mlp_g    = (const float*)d_in[14];
  const float* mlp_b    = (const float*)d_in[15];
  const float* fc1_w    = (const float*)d_in[16];
  const float* fc1_bias = (const float*)d_in[17];
  const float* fc1_g    = (const float*)d_in[18];
  const float* fc1_b    = (const float*)d_in[19];
  const float* fc2_w    = (const float*)d_in[20];
  const float* fc2_bias = (const float*)d_in[21];
  const float* fc2_g    = (const float*)d_in[22];
  const float* fc2_b    = (const float*)d_in[23];
  const float* fc3_w    = (const float*)d_in[24];
  const float* fc3_bias = (const float*)d_in[25];
  (void)in_sizes; (void)n_in; (void)out_size; (void)ws_size;

  char* ws = (char*)d_ws;
  size_t off = 0;
  auto carve = [&](size_t bytes)->char*{
    char* p = ws + off;
    off += (bytes + 255) & ~(size_t)255;
    return p;
  };
  float*     pts_t = (float*)carve((size_t)B_ * N_ * 3 * 4);     // (B,N,3)
  float*     feat  = (float*)carve((size_t)B_ * N_ * 512 * 4);   // (B,N,512) concat slices
  int*       idxb  = (int*)  carve((size_t)B_ * N_ * KNN_ * 4);
  int*       maxb  = (int*)  carve((size_t)B_ * 1024 * 4);
  float*     sumb  = (float*)carve((size_t)B_ * 1024 * 4);
  _Float16*  wp1   = (_Float16*)carve((size_t)64   * 32  * 2);
  _Float16*  wp2   = (_Float16*)carve((size_t)64   * 128 * 2);
  _Float16*  wp3   = (_Float16*)carve((size_t)128  * 128 * 2);
  _Float16*  wp4   = (_Float16*)carve((size_t)256  * 256 * 2);
  _Float16*  wpm   = (_Float16*)carve((size_t)1024 * 512 * 2);

  // pre-swizzle weights to f16 fragment-major
  pack_a_kernel<<<(64*32    + 255)/256, 256, 0, stream>>>(ec1_w, 64,   6,   32,  wp1);
  pack_a_kernel<<<(64*128   + 255)/256, 256, 0, stream>>>(ec2_w, 64,   128, 128, wp2);
  pack_a_kernel<<<(128*128  + 255)/256, 256, 0, stream>>>(ec3_w, 128,  128, 128, wp3);
  pack_a_kernel<<<(256*256  + 255)/256, 256, 0, stream>>>(ec4_w, 256,  256, 256, wp4);
  pack_a_kernel<<<(1024*512 + 255)/256, 256, 0, stream>>>(mlp_w, 1024, 512, 512, wpm);

  transpose_points_kernel<<<(B_ * N_ + 255) / 256, 256, 0, stream>>>(points, pts_t);

  dim3 gknn(N_ / 8, B_), gec(N_ / 16, B_);

  knn_kernel<<<gknn, 256, 0, stream>>>(pts_t, 3, 3, idxb);
  edgeconv_kernel<3, 64, 32><<<gec, 256, 320*32*2 + 16*64*4, stream>>>(
      pts_t, 3, idxb, wp1, ec1_g, ec1_b, feat + 0, 512);

  knn_kernel<<<gknn, 256, 0, stream>>>(feat + 0, 512, 64, idxb);
  edgeconv_kernel<64, 64, 128><<<gec, 256, 320*128*2 + 16*64*4, stream>>>(
      feat + 0, 512, idxb, wp2, ec2_g, ec2_b, feat + 64, 512);

  knn_kernel<<<gknn, 256, 0, stream>>>(feat + 64, 512, 64, idxb);
  edgeconv_kernel<64, 128, 128><<<gec, 256, 320*128*2 + 16*128*4, stream>>>(
      feat + 64, 512, idxb, wp3, ec3_g, ec3_b, feat + 128, 512);

  knn_kernel<<<gknn, 256, 0, stream>>>(feat + 128, 512, 128, idxb);
  edgeconv_kernel<128, 256, 256><<<gec, 256, 320*256*2 + 16*256*4, stream>>>(
      feat + 128, 512, idxb, wp4, ec4_g, ec4_b, feat + 256, 512);

  init_pool_kernel<<<(B_ * 1024 + 255) / 256, 256, 0, stream>>>(maxb, sumb);
  mlp_pool_kernel<<<gec, 256, 0, stream>>>(feat, wpm, mlp_g, mlp_b, maxb, sumb);
  head_kernel<<<1, 512, 0, stream>>>(maxb, sumb,
      fc1_w, fc1_bias, fc1_g, fc1_b,
      fc2_w, fc2_bias, fc2_g, fc2_b,
      fc3_w, fc3_bias, (float*)d_out);
}
